// MultiHeadedAttention_23416161697797
// MI455X (gfx1250) — matrix operand order repaired and checked by reading, not compiled, not run on hardware
//
#include <hip/hip_runtime.h>

// ---------------------------------------------------------------------------
// MI455X (gfx1250) fused MHA: QKV-proj + RoPE -> flash attention -> out-proj
// All GEMMs via v_wmma_f32_16x16x32_f16 (f16 in, f32 accumulate), wave32.
// K-tile staging in the attention kernel uses GLOBAL_LOAD_ASYNC_TO_LDS_B128.
// GEMM waves own 32x64 output (2 A-frags x 4 B-frags -> 8 WMMAs per k-step).
// ---------------------------------------------------------------------------

#define B_   2
#define S_   2048
#define D_   1024
#define H_   16
#define HD_  64
#define ROWS (B_ * S_)          // 4096
#define D3   (3 * D_)           // 3072

typedef __attribute__((ext_vector_type(16))) _Float16 v16h;
typedef __attribute__((ext_vector_type(8)))  float    v8f;

union Frag16 { v16h v; unsigned int u[8]; };

static __device__ __forceinline__ v8f wmma_f16(const Frag16& a, const Frag16& b, v8f c) {
  return __builtin_amdgcn_wmma_f32_16x16x32_f16(
      /*neg_a=*/false, a.v, /*neg_b=*/false, b.v,
      /*c_mod=*/(short)0, c, /*reuse_a=*/false, /*reuse_b=*/false);
}

static __device__ __forceinline__ unsigned short f2h(float f) {
  _Float16 h = (_Float16)f;
  return __builtin_bit_cast(unsigned short, h);
}
static __device__ __forceinline__ unsigned int pk2h(float lo, float hi) {
  return (unsigned int)f2h(lo) | ((unsigned int)f2h(hi) << 16);
}

// A-fragment K index pattern for 16-bit 16x32 A, per VGPR j and
// lane half kh: kh*16 + 2j (contiguous, matches B fragment reads).
static __device__ __forceinline__ int a_kidx(int j, int kh) {
  return kh * 16 + 2 * j;
}

// cross-lane row reductions within a 16-lane half (C/D rows live in one half)
static __device__ __forceinline__ float rmax16(float v) {
#pragma unroll
  for (int m = 1; m < 16; m <<= 1) v = fmaxf(v, __shfl_xor(v, m, 32));
  return v;
}
static __device__ __forceinline__ float rsum16(float v) {
#pragma unroll
  for (int m = 1; m < 16; m <<= 1) v += __shfl_xor(v, m, 32);
  return v;
}

// ---------------------------------------------------------------------------
// Kernel 1: QKV = X @ Win, fused RoPE on Q/K, f16 output in [B,H,S,HD].
// Block = 128 threads (4 waves); block tile = 128 rows x 64 cols; each wave
// owns 32 rows x 64 cols (one head of one of {q,k,v} per column span).
// ---------------------------------------------------------------------------
__global__ __launch_bounds__(128) void qkv_rope_kernel(
    const float* __restrict__ X, const float* __restrict__ sinq,
    const float* __restrict__ cosq, const float* __restrict__ sink,
    const float* __restrict__ cosk, const float* __restrict__ Win,
    unsigned short* __restrict__ Qb, unsigned short* __restrict__ Kb,
    unsigned short* __restrict__ Vb) {
  __shared__ unsigned short lds_a[128 * 32];  // [row][k]  f16
  __shared__ unsigned short lds_b[64 * 32];   // [n][k]    f16 (N-major == B^T staging)

  const int t = threadIdx.x;
  const int lane = t & 31, wv = t >> 5;
  const int rbase = blockIdx.y * 128;
  const int cbase = blockIdx.x * 64;

  v8f z = {};
  v8f acc[2][4];
#pragma unroll
  for (int s = 0; s < 2; ++s)
#pragma unroll
    for (int i = 0; i < 4; ++i) acc[s][i] = z;

  for (int kk = 0; kk < D_; kk += 32) {
    // ---- load phase: cluster the b128 global loads, then convert+store ----
    float4 fa[8], fb[4];
#pragma unroll
    for (int i = 0; i < 8; ++i) {               // A tile 128x32 f32, 8 float4/thread
      int e = (t + i * 128) * 4;
      int r = e >> 5, c = e & 31;
      fa[i] = *(const float4*)(X + (size_t)(rbase + r) * D_ + kk + c);
    }
#pragma unroll
    for (int i = 0; i < 4; ++i) {               // B tile 32x64 f32, coalesced along n
      int e = (t + i * 128) * 4;
      int n = e & 63, k = e >> 6;
      fb[i] = *(const float4*)(Win + (size_t)(kk + k) * D3 + cbase + n);
    }
    if (kk + 32 < D_) {                          // next-tile prefetch hints
      __builtin_prefetch(X + (size_t)(rbase + t) * D_ + kk + 32, 0, 3);
      __builtin_prefetch(Win + (size_t)(kk + 32 + (t >> 2)) * D3 + cbase, 0, 3);
    }
#pragma unroll
    for (int i = 0; i < 8; ++i) {               // A: pack 4 f16 -> one b64 DS store
      int e = (t + i * 128) * 4;
      int r = e >> 5, c = e & 31;
      uint2 p; p.x = pk2h(fa[i].x, fa[i].y); p.y = pk2h(fa[i].z, fa[i].w);
      *(uint2*)(lds_a + r * 32 + c) = p;
    }
#pragma unroll
    for (int i = 0; i < 4; ++i) {               // B: transpose into [n][k]
      int e = (t + i * 128) * 4;
      int n = e & 63, k = e >> 6;
      lds_b[(n + 0) * 32 + k] = f2h(fb[i].x);
      lds_b[(n + 1) * 32 + k] = f2h(fb[i].y);
      lds_b[(n + 2) * 32 + k] = f2h(fb[i].z);
      lds_b[(n + 3) * 32 + k] = f2h(fb[i].w);
    }
    __syncthreads();

    const int mr = lane & 15, kh = lane >> 4;
    Frag16 a[2];
#pragma unroll
    for (int s = 0; s < 2; ++s)
#pragma unroll
      for (int j = 0; j < 8; ++j)
        a[s].u[j] = *(const unsigned int*)(lds_a + (wv * 32 + s * 16 + mr) * 32 + a_kidx(j, kh));
#pragma unroll
    for (int nt = 0; nt < 4; ++nt) {
      Frag16 b;
#pragma unroll
      for (int j = 0; j < 8; ++j)
        b.u[j] = *(const unsigned int*)(lds_b + (nt * 16 + mr) * 32 + kh * 16 + 2 * j);
      acc[0][nt] = wmma_f16(a[0], b, acc[0][nt]);   // B frag reused across both
      acc[1][nt] = wmma_f16(a[1], b, acc[1][nt]);   // row subtiles
    }
    __syncthreads();
  }

  // epilogue: RoPE (q,k) or passthrough (v); D-frag: m = j + 8*(lane>>4), n = lane&15
  const int mr = lane & 15, hi = lane >> 4;
  const int sec = cbase >> 10;            // 0=q 1=k 2=v
  const int h = (cbase & (D_ - 1)) >> 6;  // head index
#pragma unroll
  for (int s2 = 0; s2 < 2; ++s2) {
#pragma unroll
    for (int j = 0; j < 8; ++j) {
      int grow = rbase + wv * 32 + s2 * 16 + j + 8 * hi;
      int b = grow >> 11, s = grow & (S_ - 1);
      size_t obase = ((size_t)(b * H_ + h) * S_ + s) * HD_;
      if (sec == 2) {
#pragma unroll
        for (int nt = 0; nt < 4; ++nt)
          Vb[obase + nt * 16 + mr] = f2h(acc[s2][nt][j]);
      } else {
        const float* sn = (sec == 0) ? sinq : sink;
        const float* cs = (sec == 0) ? cosq : cosk;
        unsigned short* ob = (sec == 0) ? Qb : Kb;
#pragma unroll
        for (int nt = 0; nt < 4; ++nt) {
          int hd = nt * 16 + mr;
          float cv = cs[s * HD_ + hd];
          float sv = sn[s * HD_ + hd];
          float partner = acc[s2][(nt + 2) & 3][j];       // hd +/- 32: same lane!
          float val = acc[s2][nt][j] * cv + ((nt < 2) ? -partner : partner) * sv;
          ob[obase + hd] = f2h(val);
        }
      }
    }
  }
}

// ---------------------------------------------------------------------------
// Kernel 2: causal flash attention per (b, h, 64-q-row tile). 4 waves, each
// wave owns 16 q rows and a private LDS slice (no block barriers: waves have
// different causal trip counts). Per 32-key tile: 4 score WMMAs + 4 PV WMMAs.
// K tile staged via async global->LDS DMA (ASYNCcnt), V staged transposed.
// ---------------------------------------------------------------------------
__global__ __launch_bounds__(128) void attn_kernel(
    const unsigned short* __restrict__ Qb, const unsigned short* __restrict__ Kb,
    const unsigned short* __restrict__ Vb, unsigned short* __restrict__ AO) {
  __shared__ unsigned short lds[4 * 4608];
  const int lane = threadIdx.x & 31;
  const int wv = threadIdx.x >> 5;
  unsigned short* lds_k  = lds + wv * 4608;   // [32 keys][64 hd]  (== N x K for K^T)
  unsigned short* lds_vt = lds_k + 2048;      // [64 hd][32 keys]  (N x K for V)
  unsigned short* lds_p  = lds_vt + 2048;     // [16 q][32 keys]

  const int qt = blockIdx.x, h = blockIdx.y, b = blockIdx.z;
  const int qbase = qt * 64 + wv * 16;
  const size_t headoff = (size_t)(b * H_ + h) * S_ * HD_;
  const unsigned short* Qh = Qb + headoff;
  const unsigned short* Kh = Kb + headoff;
  const unsigned short* Vh = Vb + headoff;

  const int mr = lane & 15, kh = lane >> 4;

  // Q A-fragments (16 x 64 => two K=32 chunks), loaded straight from global
  Frag16 qa[2];
#pragma unroll
  for (int c = 0; c < 2; ++c)
#pragma unroll
    for (int j = 0; j < 8; ++j)
      qa[c].u[j] = *(const unsigned int*)(Qh + (size_t)(qbase + mr) * HD_ + c * 32 + a_kidx(j, kh));

  v8f z = {};
  v8f acc[4];
#pragma unroll
  for (int i = 0; i < 4; ++i) acc[i] = z;
  float mrun[8], lrun[8];
#pragma unroll
  for (int j = 0; j < 8; ++j) { mrun[j] = -3.0e38f; lrun[j] = 0.0f; }

  // LDS byte offset of this wave's K slice (flat LDS addr keeps offset in [31:0])
  const unsigned int lds_k_base = (unsigned int)(size_t)(lds_k + lane * 64);

  const int ktmax = (qbase + 15) >> 5;
  for (int kt = 0; kt <= ktmax; ++kt) {
    const int kb0 = kt * 32;
    // ---- K rows: async global->LDS copy, one b128 per 16 bytes per lane ----
    {
      unsigned long long gK =
          (unsigned long long)(size_t)(Kh + (size_t)(kb0 + lane) * HD_);
#pragma unroll
      for (int c = 0; c < 8; ++c) {
        unsigned int la = lds_k_base + c * 16;
        unsigned long long ga = gK + c * 16;
        asm volatile("global_load_async_to_lds_b128 %0, %1, off"
                     :: "v"(la), "v"(ga) : "memory");
      }
    }
    // ---- V rows: vector loads + transposed b16 scatter into [hd][key] ----
    {
      const uint4* vs = (const uint4*)(Vh + (size_t)(kb0 + lane) * HD_);
      uint4 vq[8];
#pragma unroll
      for (int c = 0; c < 8; ++c) vq[c] = vs[c];
#pragma unroll
      for (int c = 0; c < 8; ++c) {
#pragma unroll
        for (int e = 0; e < 8; ++e) {
          unsigned int w = (e < 2) ? vq[c].x : (e < 4) ? vq[c].y : (e < 6) ? vq[c].z : vq[c].w;
          lds_vt[(c * 8 + e) * 32 + lane] = (unsigned short)(w >> ((e & 1) * 16));
        }
      }
    }
    if (kb0 + 32 <= qbase + 15) {  // prefetch next tile's K/V rows
      __builtin_prefetch(Kh + (size_t)(kb0 + 32 + lane) * HD_, 0, 3);
      __builtin_prefetch(Vh + (size_t)(kb0 + 32 + lane) * HD_, 0, 3);
    }
    asm volatile("s_wait_asynccnt 0x0" ::: "memory");   // K tile landed in LDS
    asm volatile("s_wait_dscnt 0x0" ::: "memory");      // V transpose visible

    // scores: 16 q x 32 keys (two 16x16 D-frags), K-dim 64 accumulated
    v8f sf[2];
#pragma unroll
    for (int k2 = 0; k2 < 2; ++k2) {
      v8f s = z;
#pragma unroll
      for (int c = 0; c < 2; ++c) {
        Frag16 bk;
#pragma unroll
        for (int j = 0; j < 8; ++j)
          bk.u[j] = *(const unsigned int*)(lds_k + (k2 * 16 + mr) * 64 + c * 32 + kh * 16 + 2 * j);
        s = wmma_f16(qa[c], bk, s);
      }
      sf[k2] = s;
    }
    // scale + causal mask (D-frag: row = j + 8*kh, col = mr)
#pragma unroll
    for (int k2 = 0; k2 < 2; ++k2)
#pragma unroll
      for (int j = 0; j < 8; ++j) {
        float v = sf[k2][j] * 0.125f;
        int kg = kb0 + k2 * 16 + mr;
        int qg = qbase + j + 8 * kh;
        sf[k2][j] = (kg <= qg) ? v : -3.0e38f;
      }
    // online softmax update over the 32-key tile
    float pr0[8], pr1[8];
#pragma unroll
    for (int j = 0; j < 8; ++j) {
      float rm = rmax16(fmaxf(sf[0][j], sf[1][j]));
      float nm = fmaxf(mrun[j], rm);
      float corr = __expf(mrun[j] - nm);
      mrun[j] = nm;
      float p0 = __expf(sf[0][j] - nm);
      float p1 = __expf(sf[1][j] - nm);
      lrun[j] = lrun[j] * corr + rsum16(p0 + p1);
      acc[0][j] *= corr; acc[1][j] *= corr; acc[2][j] *= corr; acc[3][j] *= corr;
      pr0[j] = p0; pr1[j] = p1;
    }
    // P: D-layout -> LDS -> A-layout
#pragma unroll
    for (int j = 0; j < 8; ++j) {
      int row = j + 8 * kh;
      lds_p[row * 32 + mr]      = f2h(pr0[j]);
      lds_p[row * 32 + 16 + mr] = f2h(pr1[j]);
    }
    asm volatile("s_wait_dscnt 0x0" ::: "memory");
    Frag16 pa;
#pragma unroll
    for (int j = 0; j < 8; ++j)
      pa.u[j] = *(const unsigned int*)(lds_p + mr * 32 + a_kidx(j, kh));
    // O += P @ V  (4 hd tiles of 16)
#pragma unroll
    for (int nt = 0; nt < 4; ++nt) {
      Frag16 bv;
#pragma unroll
      for (int j = 0; j < 8; ++j)
        bv.u[j] = *(const unsigned int*)(lds_vt + (nt * 16 + mr) * 32 + kh * 16 + 2 * j);
      acc[nt] = wmma_f16(pa, bv, acc[nt]);
    }
  }

  // finalize: divide by l, store f16 to [B,S,D] for the out-proj GEMM
#pragma unroll
  for (int nt = 0; nt < 4; ++nt)
#pragma unroll
    for (int j = 0; j < 8; ++j) {
      int s = qbase + j + 8 * kh;
      int d = h * HD_ + nt * 16 + mr;
      AO[((size_t)(b * S_ + s)) * D_ + d] = f2h(acc[nt][j] / lrun[j]);
    }
}

// ---------------------------------------------------------------------------
// Kernel 3: Out = AO(f16) @ Wout, f32 result. Same 32x64-per-wave tiling.
// ---------------------------------------------------------------------------
__global__ __launch_bounds__(128) void oproj_kernel(
    const unsigned short* __restrict__ AO, const float* __restrict__ Wout,
    float* __restrict__ Out) {
  __shared__ unsigned short lds_a[128 * 32];
  __shared__ unsigned short lds_b[64 * 32];
  const int t = threadIdx.x;
  const int lane = t & 31, wv = t >> 5;
  const int rbase = blockIdx.y * 128;
  const int cbase = blockIdx.x * 64;

  v8f z = {};
  v8f acc[2][4];
#pragma unroll
  for (int s = 0; s < 2; ++s)
#pragma unroll
    for (int i = 0; i < 4; ++i) acc[s][i] = z;

  for (int kk = 0; kk < D_; kk += 32) {
    // A tile: raw f16 rows, uint4 copies (4 per thread); B: f32 -> f16 transpose
    uint4 ua[4]; float4 fb[4];
#pragma unroll
    for (int i = 0; i < 4; ++i) {
      int e = (t + i * 128) * 8;
      int r = e >> 5, c = e & 31;
      ua[i] = *(const uint4*)(AO + (size_t)(rbase + r) * D_ + kk + c);
    }
#pragma unroll
    for (int i = 0; i < 4; ++i) {
      int e = (t + i * 128) * 4;
      int n = e & 63, k = e >> 6;
      fb[i] = *(const float4*)(Wout + (size_t)(kk + k) * D_ + cbase + n);
    }
    if (kk + 32 < D_)
      __builtin_prefetch(Wout + (size_t)(kk + 32 + (t >> 2)) * D_ + cbase, 0, 3);
#pragma unroll
    for (int i = 0; i < 4; ++i) {
      int e = (t + i * 128) * 8;
      int r = e >> 5, c = e & 31;
      *(uint4*)(lds_a + r * 32 + c) = ua[i];
    }
#pragma unroll
    for (int i = 0; i < 4; ++i) {
      int e = (t + i * 128) * 4;
      int n = e & 63, k = e >> 6;
      lds_b[(n + 0) * 32 + k] = f2h(fb[i].x);
      lds_b[(n + 1) * 32 + k] = f2h(fb[i].y);
      lds_b[(n + 2) * 32 + k] = f2h(fb[i].z);
      lds_b[(n + 3) * 32 + k] = f2h(fb[i].w);
    }
    __syncthreads();

    const int mr = lane & 15, kh = lane >> 4;
    Frag16 a[2];
#pragma unroll
    for (int s = 0; s < 2; ++s)
#pragma unroll
      for (int j = 0; j < 8; ++j)
        a[s].u[j] = *(const unsigned int*)(lds_a + (wv * 32 + s * 16 + mr) * 32 + a_kidx(j, kh));
#pragma unroll
    for (int nt = 0; nt < 4; ++nt) {
      Frag16 b;
#pragma unroll
      for (int j = 0; j < 8; ++j)
        b.u[j] = *(const unsigned int*)(lds_b + (nt * 16 + mr) * 32 + kh * 16 + 2 * j);
      acc[0][nt] = wmma_f16(a[0], b, acc[0][nt]);
      acc[1][nt] = wmma_f16(a[1], b, acc[1][nt]);
    }
    __syncthreads();
  }

  const int mr = lane & 15, hi = lane >> 4;
#pragma unroll
  for (int s = 0; s < 2; ++s)
#pragma unroll
    for (int nt = 0; nt < 4; ++nt)
#pragma unroll
      for (int j = 0; j < 8; ++j) {
        int grow = rbase + wv * 32 + s * 16 + j + 8 * hi;
        Out[(size_t)grow * D_ + cbase + nt * 16 + mr] = acc[s][nt][j];
      }
}

// ---------------------------------------------------------------------------
extern "C" void kernel_launch(void* const* d_in, const int* in_sizes, int n_in,
                              void* d_out, int out_size, void* d_ws, size_t ws_size,
                              hipStream_t stream) {
  (void)in_sizes; (void)n_in; (void)out_size; (void)ws_size;
  const float* X    = (const float*)d_in[0];
  const float* sinq = (const float*)d_in[1];
  const float* cosq = (const float*)d_in[2];
  const float* sink = (const float*)d_in[3];
  const float* cosk = (const float*)d_in[4];
  const float* Win  = (const float*)d_in[5];
  const float* Wout = (const float*)d_in[6];

  const size_t NBHSD = (size_t)B_ * H_ * S_ * HD_;   // 4M elems per buffer
  unsigned short* Qb = (unsigned short*)d_ws;
  unsigned short* Kb = Qb + NBHSD;
  unsigned short* Vb = Kb + NBHSD;
  unsigned short* AO = Vb + NBHSD;

  qkv_rope_kernel<<<dim3(D3 / 64, ROWS / 128), 128, 0, stream>>>(
      X, sinq, cosq, sink, cosk, Win, Qb, Kb, Vb);
  attn_kernel<<<dim3(S_ / 64, H_, B_), 128, 0, stream>>>(Qb, Kb, Vb, AO);
  oproj_kernel<<<dim3(D_ / 64, ROWS / 128), 128, 0, stream>>>(AO, Wout, (float*)d_out);
}
